// DLRM_76450417869475
// MI455X (gfx1250) — compile-verified, matrix operand order checked by reading, not run on hardware
//
#include <hip/hip_runtime.h>

// ---------------- problem constants ----------------
#define BATCH 16384
#define DDIM  13
#define EDIM  64
#define FCAT  26
#define VOCAB 200000
#define NI    27          // F + 1
#define NPAIR 351         // 27*26/2
#define XPADK 416         // 64 + 351 padded to 13*32
#define KPAD0 32          // 13 padded to 32
#define MAXK  512         // largest GEMM K
#define LDSPAD 8          // halves of padding per LDS row (16B -> 4-bank skew)

typedef __attribute__((ext_vector_type(16))) _Float16 v16h;
typedef __attribute__((ext_vector_type(8)))  _Float16 v8h;
typedef __attribute__((ext_vector_type(8)))  float    v8f;
typedef __attribute__((ext_vector_type(4)))  float    v4f;

#define SHUF16(lo, hi) __builtin_shufflevector(lo, hi, 0,1,2,3,4,5,6,7,8,9,10,11,12,13,14,15)

// ---------------- prep kernels ----------------
__global__ __launch_bounds__(256) void cvt_pad_dense(const float* __restrict__ d,
                                                     _Float16* __restrict__ xd) {
    int i = blockIdx.x * 256 + threadIdx.x;           // BATCH*32 elements
    if (i >= BATCH * KPAD0) return;
    int b = i >> 5, k = i & 31;
    xd[i] = (k < DDIM) ? (_Float16)d[b * DDIM + k] : (_Float16)0.0f;
}

// W is (Korig x N) row-major f32 -> Wt is (N x ldk) row-major f16, K zero-padded
__global__ __launch_bounds__(256) void transpose_cvt(const float* __restrict__ W,
                                                     _Float16* __restrict__ Wt,
                                                     int N, int Korig, int ldk) {
    int i = blockIdx.x * 256 + threadIdx.x;
    if (i >= N * ldk) return;
    int n = i / ldk, k = i % ldk;
    Wt[i] = (k < Korig) ? (_Float16)W[k * N + n] : (_Float16)0.0f;
}

__global__ __launch_bounds__(256) void zero_lastcol(_Float16* __restrict__ xpad) {
    int b = blockIdx.x * 256 + threadIdx.x;
    if (b < BATCH) xpad[(size_t)b * XPADK + (XPADK - 1)] = (_Float16)0.0f;
}

// ---------------- WMMA GEMM with LDS-shared weight tile ----------------
// C[M x N] = act(A[M x K] * Wt^T + bias).  A row-major f16 (lda = K, mult of 32),
// Wt (N x K) row-major f16.  Block = 8 consecutive M-tiles x ONE N-tile: all 8
// waves share the same 16xK weight fragment, staged once in LDS (padded rows ->
// conflict-free ds_load_b128).  grid = (M/16/8) * Ntiles blocks, 256 threads.
__global__ __launch_bounds__(256) void gemm16_f16(const _Float16* __restrict__ A, int lda,
                                                  const _Float16* __restrict__ Wt, int ldw,
                                                  const float* __restrict__ bias,
                                                  _Float16* __restrict__ C, int ldc,
                                                  int K, int Ntiles, int relu) {
    __shared__ __align__(16) _Float16 bs[16 * (MAXK + LDSPAD)];
    const int SK = K + LDSPAD;                         // LDS row stride (halves)

    int lane = threadIdx.x & 31;
    int wave = threadIdx.x >> 5;
    int bm   = blockIdx.x / Ntiles;                    // group of 8 M-tiles
    int nt   = blockIdx.x % Ntiles;

    // cooperative stage: Wt rows [nt*16, nt*16+16) x K -> LDS (16B chunks)
    int chunksPerRow = K >> 3;                         // 8-half chunks
    for (int c = threadIdx.x; c < 16 * chunksPerRow; c += 256) {
        int row = c / chunksPerRow;
        int off = (c - row * chunksPerRow) << 3;
        *(v8h*)(bs + row * SK + off) =
            *(const v8h*)(Wt + (size_t)(nt * 16 + row) * ldw + off);
    }
    __syncthreads();

    int rsel = lane & 15, half = lane >> 4;
    int mt = bm * 8 + wave;
    const _Float16* arow = A  + (size_t)(mt * 16 + rsel) * lda + half * 8;
    const _Float16* brow = bs + rsel * SK + half * 8;

    v8f c = {};
    for (int kk = 0; kk < K; kk += 32) {
        v8h a0 = *(const v8h*)(arow + kk);
        v8h a1 = *(const v8h*)(arow + kk + 16);
        v8h b0 = *(const v8h*)(brow + kk);             // ds_load_b128
        v8h b1 = *(const v8h*)(brow + kk + 16);
        if (kk + 32 < K)                               // gfx1250 global_prefetch_b8
            __builtin_prefetch(arow + kk + 32, 0, 1);
        v16h a = SHUF16(a0, a1);
        v16h b = SHUF16(b0, b1);
        c = __builtin_amdgcn_wmma_f32_16x16x32_f16(false, a, false, b, (short)0, c,
                                                   false, false);
    }

    float bv = bias[nt * 16 + rsel];
    _Float16* cp = C + (size_t)(mt * 16 + half * 8) * ldc + nt * 16 + rsel;
#pragma unroll
    for (int r = 0; r < 8; ++r) {
        float v = c[r] + bv;
        if (relu) v = fmaxf(v, 0.0f);
        cp[(size_t)r * ldc] = (_Float16)v;
    }
}

// ---------------- interaction: Z = T*T^T, scatter upper triangle ----------------
__device__ __forceinline__ v8h load_T_row8(const _Float16* __restrict__ xrow,
                                           const int* __restrict__ cat,
                                           const float* __restrict__ tables,
                                           int b, int r, int ko) {
    v8h out = {};
    if (r == 0) {
        out = *(const v8h*)(xrow + ko);               // dense_out (already f16)
    } else if (r <= FCAT) {
        int f = r - 1;
        int idx = cat[f * BATCH + b];
        const float* src = tables + ((size_t)f * VOCAB + (size_t)idx) * EDIM + ko;
        v4f lo = *(const v4f*)(src);
        v4f hi = *(const v4f*)(src + 4);
#pragma unroll
        for (int i = 0; i < 4; ++i) {
            out[i]     = (_Float16)lo[i];
            out[i + 4] = (_Float16)hi[i];
        }
    }
    return out;                                       // rows 27..31: zeros (pad)
}

__global__ __launch_bounds__(256) void interact_kernel(const _Float16* __restrict__ xpad,
                                                       const int* __restrict__ cat,
                                                       const float* __restrict__ tables,
                                                       _Float16* __restrict__ xout) {
    int lane = threadIdx.x & 31;
    int wv = (blockIdx.x * 256 + threadIdx.x) >> 5;   // BATCH*3 waves
    int b = wv / 3, t = wv - 3 * b;
    int mt = (t == 2) ? 1 : 0;                        // tiles (0,0) (0,1) (1,1)
    int nt = (t == 0) ? 0 : 1;
    int rsel = lane & 15, half = lane >> 4;
    const _Float16* xrow = xpad + (size_t)b * XPADK;
    int ra = mt * 16 + rsel, rb = nt * 16 + rsel;

    v8f c = {};
#pragma unroll
    for (int kk = 0; kk < EDIM; kk += 32) {
        v8h a0 = load_T_row8(xrow, cat, tables, b, ra, kk + half * 8);
        v8h a1 = load_T_row8(xrow, cat, tables, b, ra, kk + 16 + half * 8);
        v8h b0 = load_T_row8(xrow, cat, tables, b, rb, kk + half * 8);
        v8h b1 = load_T_row8(xrow, cat, tables, b, rb, kk + 16 + half * 8);
        v16h a = SHUF16(a0, a1);
        v16h bm = SHUF16(b0, b1);
        c = __builtin_amdgcn_wmma_f32_16x16x32_f16(false, a, false, bm, (short)0, c,
                                                   false, false);
    }

    int Cc = nt * 16 + rsel;
#pragma unroll
    for (int r = 0; r < 8; ++r) {
        int R = mt * 16 + half * 8 + r;
        if (R < NI && Cc < NI && Cc > R) {
            int p = R * (NI - 1) - (R * (R - 1)) / 2 + (Cc - R - 1);  // triu k=1 order
            xout[(size_t)b * XPADK + EDIM + p] = (_Float16)c[r];
        }
    }
}

// ---------------- head: dot(h2, w_t2) + b, sigmoid * 5 ----------------
__global__ __launch_bounds__(256) void final_head(const _Float16* __restrict__ h2,
                                                  const float* __restrict__ w,
                                                  const float* __restrict__ b,
                                                  float* __restrict__ out) {
    int lane = threadIdx.x & 31;
    int row = (blockIdx.x * 256 + threadIdx.x) >> 5;
    const _Float16* hr = h2 + (size_t)row * 256;
    float s = 0.0f;
#pragma unroll
    for (int i = 0; i < 8; ++i) s += (float)hr[lane + 32 * i] * w[lane + 32 * i];
#pragma unroll
    for (int off = 16; off > 0; off >>= 1) s += __shfl_down(s, off, 32);
    if (lane == 0) {
        float z = s + b[0];
        out[row] = 5.0f / (1.0f + __expf(-z));
    }
}

// ---------------- host launch ----------------
extern "C" void kernel_launch(void* const* d_in, const int* in_sizes, int n_in,
                              void* d_out, int out_size, void* d_ws, size_t ws_size,
                              hipStream_t stream) {
    const float* dense  = (const float*)d_in[0];
    const int*   cat    = (const int*)d_in[1];
    const float* tables = (const float*)d_in[2];
    const float* w_b0 = (const float*)d_in[3];
    const float* b_b0 = (const float*)d_in[4];
    const float* w_b1 = (const float*)d_in[5];
    const float* b_b1 = (const float*)d_in[6];
    const float* w_t0 = (const float*)d_in[7];
    const float* b_t0 = (const float*)d_in[8];
    const float* w_t1 = (const float*)d_in[9];
    const float* b_t1 = (const float*)d_in[10];
    const float* w_t2 = (const float*)d_in[11];
    const float* b_t2 = (const float*)d_in[12];
    float* out = (float*)d_out;

    char* ws = (char*)d_ws;
    size_t o = 0;
    auto take = [&](size_t bytes) { size_t r = o; o = (o + bytes + 255) & ~(size_t)255; return r; };
    _Float16* xd   = (_Float16*)(ws + take((size_t)BATCH * KPAD0 * 2));
    _Float16* wb0t = (_Float16*)(ws + take((size_t)512 * KPAD0 * 2));
    _Float16* wb1t = (_Float16*)(ws + take((size_t)64 * 512 * 2));
    _Float16* wt0t = (_Float16*)(ws + take((size_t)512 * XPADK * 2));
    _Float16* wt1t = (_Float16*)(ws + take((size_t)256 * 512 * 2));
    _Float16* h0   = (_Float16*)(ws + take((size_t)BATCH * 512 * 2));
    _Float16* xpad = (_Float16*)(ws + take((size_t)BATCH * XPADK * 2));
    _Float16* h1   = (_Float16*)(ws + take((size_t)BATCH * 512 * 2));
    _Float16* h2   = (_Float16*)(ws + take((size_t)BATCH * 256 * 2));

    // prep: conversions / transposes / padding
    cvt_pad_dense<<<(BATCH * KPAD0 + 255) / 256, 256, 0, stream>>>(dense, xd);
    transpose_cvt<<<(512 * KPAD0 + 255) / 256, 256, 0, stream>>>(w_b0, wb0t, 512, DDIM, KPAD0);
    transpose_cvt<<<(64 * 512 + 255) / 256, 256, 0, stream>>>(w_b1, wb1t, 64, 512, 512);
    transpose_cvt<<<(512 * XPADK + 255) / 256, 256, 0, stream>>>(w_t0, wt0t, 512, 415, XPADK);
    transpose_cvt<<<(256 * 512 + 255) / 256, 256, 0, stream>>>(w_t1, wt1t, 256, 512, 512);
    zero_lastcol<<<(BATCH + 255) / 256, 256, 0, stream>>>(xpad);

    const int MG = BATCH / 16 / 8;   // 128 groups of 8 M-tiles
    // bottom MLP: (B x 32) @ (32 x 512) -> relu -> h0
    gemm16_f16<<<MG * 32, 256, 0, stream>>>(xd, KPAD0, wb0t, KPAD0, b_b0, h0, 512,
                                            KPAD0, 32, 1);
    // (B x 512) @ (512 x 64) -> dense_out into xpad cols [0,64)
    gemm16_f16<<<MG * 4, 256, 0, stream>>>(h0, 512, wb1t, 512, b_b1, xpad, XPADK,
                                           512, 4, 0);
    // interaction: gather embeddings + Z = T T^T, scatter pairs into xpad cols [64,415)
    interact_kernel<<<BATCH * 3 / 8, 256, 0, stream>>>(xpad, cat, tables, xpad);
    // top MLP
    gemm16_f16<<<MG * 32, 256, 0, stream>>>(xpad, XPADK, wt0t, XPADK, b_t0, h1, 512,
                                            XPADK, 32, 1);
    gemm16_f16<<<MG * 16, 256, 0, stream>>>(h1, 512, wt1t, 512, b_t1, h2, 256,
                                            512, 16, 1);
    final_head<<<BATCH / 8, 256, 0, stream>>>(h2, w_t2, b_t2, out);
}
